// MACEAdapted_72181220377112
// MI455X (gfx1250) — matrix operand order repaired and checked by reading, not compiled, not run on hardware
//
#include <hip/hip_runtime.h>
#include <hip/hip_bf16.h>

#define NNODE  64
#define HCH    128
#define NBASIS 8
#define BGRAPH 64
#define TDIM   32
#define JTILE  16
#define RMAXF  5.0f
#define AVGN   63.0f
#define PI_F   3.14159265358979323846f

typedef __attribute__((ext_vector_type(16))) __bf16 v16bf;
typedef __attribute__((ext_vector_type(8)))  __bf16 v8bf;
typedef __attribute__((ext_vector_type(8)))  float  v8f;

static __device__ __forceinline__ float silu_f(float x) {
    return x / (1.0f + __expf(-x));
}

// ---- WMMA helpers: D(16x16) += A(16x32 bf16) * B(32x16 bf16), f32 accum ----
// A tile from LDS, row-major [16][128] bf16. 16-bit A layout: lanes 0-15 hold
// K = 8h..8h+7 and 16+8h..16+8h+7 where h = lane>>4 (two contiguous 16B chunks).
static __device__ __forceinline__ v16bf ld_a_tile(const __bf16* A, int lane, int kk) {
    const int half = lane >> 4;
    const int m    = lane & 15;
    const __bf16* p = A + m * HCH + kk * 32 + 8 * half;
    v8bf c0 = *(const v8bf*)(p);
    v8bf c1 = *(const v8bf*)(p + 16);
    return __builtin_shufflevector(c0, c1, 0,1,2,3,4,5,6,7,8,9,10,11,12,13,14,15);
}

// B tile from LDS weight stored TRANSPOSED [n][k] bf16 so each lane reads two
// 16-byte runs: lane n = lane&15, K = kk*32 + 8h..8h+7 and 16+8h..16+8h+7, h = lane>>4.
static __device__ __forceinline__ v16bf ld_b_tile(const __bf16* WT, int lane, int ct, int kk) {
    const int half = lane >> 4;
    const int n    = lane & 15;
    const __bf16* p = WT + (ct * 16 + n) * HCH + kk * 32 + 8 * half;
    v8bf c0 = *(const v8bf*)(p);
    v8bf c1 = *(const v8bf*)(p + 16);
    return __builtin_shufflevector(c0, c1, 0,1,2,3,4,5,6,7,8,9,10,11,12,13,14,15);
}

// One 16x16 output tile of  [16,128] @ [128,128]  (4 k-steps of 32)
static __device__ __forceinline__ v8f wmma_row_tile(const __bf16* A, const __bf16* WT,
                                                    int lane, int ct) {
    v8f acc = {};
#pragma unroll
    for (int kk = 0; kk < 4; ++kk) {
        v16bf a = ld_a_tile(A, lane, kk);
        v16bf b = ld_b_tile(WT, lane, ct, kk);
        acc = __builtin_amdgcn_wmma_f32_16x16x32_bf16(false, a, false, b,
                                                      (short)0, acc, false, false);
    }
    return acc;
}

// Stage a [128][128] f32 global weight into LDS transposed bf16 [n][k].
static __device__ __forceinline__ void stage_wT(__bf16* WT, const float* __restrict__ Wg, int tid) {
    for (int idx = tid; idx < HCH * HCH; idx += 128) {
        const int k = idx >> 7;
        const int n = idx & 127;
        WT[n * HCH + k] = (__bf16)Wg[idx];
    }
}

// ---------------------------------------------------------------------------
// Kernel 0: h0 = emb[z] + t @ Wt ; v = 0
// ---------------------------------------------------------------------------
__global__ void mace_init_kernel(const float* __restrict__ gfeat,   // [B,T]
                                 const int*   __restrict__ z,       // [B,N]
                                 const float* __restrict__ emb,     // [S,H]
                                 const float* __restrict__ Wt,      // [T,H]
                                 float* __restrict__ hA,            // [B,N,H]
                                 float* __restrict__ vAcc)          // [B,N,H,3]
{
    const int b = blockIdx.x;
    const int c = threadIdx.x;      // 128 threads = channels
    float tw = 0.0f;
#pragma unroll
    for (int k = 0; k < TDIM; ++k)
        tw = fmaf(gfeat[b * TDIM + k], Wt[k * HCH + c], tw);
    for (int n = 0; n < NNODE; ++n) {
        const int zi = z[b * NNODE + n];
        hA[(b * NNODE + n) * HCH + c] = emb[zi * HCH + c] + tw;
    }
    for (int i = c; i < NNODE * HCH * 3; i += 128)
        vAcc[b * NNODE * HCH * 3 + i] = 0.0f;
}

// ---------------------------------------------------------------------------
// Kernel 1 (per interaction layer): message passing + node update.
// One WG = (graph b, 16-receiver tile). 128 threads = 4 wave32.
// All 64x16 edge geometries (Bessel basis + unit vectors) are precomputed in
// bulk; the sender loop is pure MLP->WMMA with ONE barrier per sender via a
// double-buffered A tile. Message accumulators a0/a1 live in registers in the
// WMMA C layout: lane (half,nl), reg r -> row m=r+8*half, col=ct*16+nl.
// ---------------------------------------------------------------------------
__global__ void __launch_bounds__(128)
mace_layer_kernel(const float* __restrict__ pos,    // [B,N,3]
                  const float* __restrict__ Wr1l,   // [8,128]
                  const float* __restrict__ Wr2l,   // [128,128]
                  const float* __restrict__ Wupdl,  // [128,128]
                  const float* __restrict__ Wmixl,  // [128,128]
                  const float* __restrict__ hIn,    // [B,N,H]
                  float* __restrict__ hOut,         // [B,N,H]
                  float* __restrict__ vAcc)         // [B,N,H,3]
{
    __shared__ __align__(32) __bf16 WT[HCH * HCH];            // 32 KB
    __shared__ __align__(32) __bf16 AbufD[2][JTILE * HCH];    // 8 KB (double buffer)
    __shared__ float Wr1s[NBASIS * HCH];                      // 4 KB
    __shared__ float posb[NNODE * 3];                         // 768 B
    __shared__ float rb_all[NNODE * JTILE * NBASIS];          // 32 KB
    __shared__ float u_all[NNODE * JTILE * 3];                // 12 KB

    const int tid  = threadIdx.x;
    const int lane = tid & 31;
    const int wv   = tid >> 5;
    const int b    = blockIdx.x >> 2;
    const int j0   = (blockIdx.x & 3) * JTILE;
    const int half = lane >> 4;
    const int nl   = lane & 15;

    // register accumulators: [col-tile][row-reg](+dim)
    float a0r[2][8];
    float a1r[2][8][3];
#pragma unroll
    for (int t = 0; t < 2; ++t)
#pragma unroll
        for (int r = 0; r < 8; ++r) {
            a0r[t][r] = 0.0f;
            a1r[t][r][0] = 0.0f; a1r[t][r][1] = 0.0f; a1r[t][r][2] = 0.0f;
        }

    for (int i = tid; i < NNODE * 3; i += 128)    posb[i] = pos[b * NNODE * 3 + i];
    for (int i = tid; i < NBASIS * HCH; i += 128) Wr1s[i] = Wr1l[i];
    stage_wT(WT, Wr2l, tid);
    __syncthreads();   // posb ready

    // ---- bulk edge geometry: all 64 senders x 16 receivers ----
    for (int e = tid; e < NNODE * JTILE; e += 128) {
        const int s  = e >> 4;         // sender
        const int jr = e & 15;         // receiver row in tile
        const int j  = j0 + jr;
        const float ex = posb[j * 3 + 0] - posb[s * 3 + 0];
        const float ey = posb[j * 3 + 1] - posb[s * 3 + 1];
        const float ez = posb[j * 3 + 2] - posb[s * 3 + 2];
        const float r  = __fsqrt_rn(ex * ex + ey * ey + ez * ez + 1e-12f);
        const float inv = 1.0f / r;
        u_all[e * 3 + 0] = ex * inv;
        u_all[e * 3 + 1] = ey * inv;
        u_all[e * 3 + 2] = ez * inv;
        const float fc = (r < RMAXF) ? 0.5f * (__cosf(PI_F * r / RMAXF) + 1.0f) : 0.0f;
        const float sc = (j == s) ? 0.0f : fc * inv;   // self-edge mask => zero rb
#pragma unroll
        for (int nb = 0; nb < NBASIS; ++nb)
            rb_all[e * NBASIS + nb] = __sinf((float)(nb + 1) * PI_F * r / RMAXF) * sc;
    }
    __syncthreads();

    // ---- sender sweep: one barrier per iteration (double-buffered A tile) ----
    for (int s = 0; s < NNODE; ++s) {
        __bf16* Ab = AbufD[s & 1];
        // hidden = silu(rb @ Wr1) -> bf16 A tile. thread = output channel.
        {
            const int c = tid;
            const float* rbs = &rb_all[s * JTILE * NBASIS];
#pragma unroll
            for (int m = 0; m < JTILE; ++m) {
                float acc = 0.0f;
#pragma unroll
                for (int nb = 0; nb < NBASIS; ++nb)
                    acc = fmaf(rbs[m * NBASIS + nb], Wr1s[nb * HCH + c], acc);
                Ab[m * HCH + c] = (__bf16)silu_f(acc);
            }
        }
        __syncthreads();

        // w = hidden @ Wr2 (WMMA); m0 = w * h[s]; register accumulate
        {
            const float* us = &u_all[s * JTILE * 3];
#pragma unroll
            for (int t = 0; t < 2; ++t) {
                const int ct = wv * 2 + t;
                v8f acc = wmma_row_tile(Ab, WT, lane, ct);
                const int ncol = ct * 16 + nl;
                const float hs = hIn[(b * NNODE + s) * HCH + ncol];
#pragma unroll
                for (int r = 0; r < 8; ++r) {
                    const int m = r + 8 * half;
                    const float f = acc[r] * hs;
                    a0r[t][r] += f;
                    a1r[t][r][0] = fmaf(f, us[m * 3 + 0], a1r[t][r][0]);
                    a1r[t][r][1] = fmaf(f, us[m * 3 + 1], a1r[t][r][1]);
                    a1r[t][r][2] = fmaf(f, us[m * 3 + 2], a1r[t][r][2]);
                }
            }
        }
        // no trailing barrier: next iteration writes the OTHER A buffer; its
        // mid-iteration barrier orders phase3(s) before phase2(s+2) reuse.
    }
    __syncthreads();

    // ---- h update: hOut = hIn + (a0/63) @ Wupd ----
    stage_wT(WT, Wupdl, tid);
    // lanes scatter their register accumulators straight into the bf16 A tile
#pragma unroll
    for (int t = 0; t < 2; ++t) {
        const int ncol = (wv * 2 + t) * 16 + nl;
#pragma unroll
        for (int r = 0; r < 8; ++r)
            AbufD[0][(r + 8 * half) * HCH + ncol] = (__bf16)(a0r[t][r] * (1.0f / AVGN));
    }
    __syncthreads();
    {
#pragma unroll
        for (int t = 0; t < 2; ++t) {
            const int ct = wv * 2 + t;
            v8f acc = wmma_row_tile(AbufD[0], WT, lane, ct);
            const int ncol = ct * 16 + nl;
#pragma unroll
            for (int r = 0; r < 8; ++r) {
                const int m  = r + 8 * half;
                const int gi = (b * NNODE + j0 + m) * HCH + ncol;
                hOut[gi] = hIn[gi] + acc[r];
            }
        }
    }
    __syncthreads();

    // ---- v update: v[...,d] += (a1[...,d]/63) @ Wmix, d = 0..2 ----
    stage_wT(WT, Wmixl, tid);
#pragma unroll
    for (int d = 0; d < 3; ++d) {
        __bf16* Ab = AbufD[d & 1];
#pragma unroll
        for (int t = 0; t < 2; ++t) {
            const int ncol = (wv * 2 + t) * 16 + nl;
#pragma unroll
            for (int r = 0; r < 8; ++r)
                Ab[(r + 8 * half) * HCH + ncol] = (__bf16)(a1r[t][r][d] * (1.0f / AVGN));
        }
        __syncthreads();
#pragma unroll
        for (int t = 0; t < 2; ++t) {
            const int ct = wv * 2 + t;
            v8f acc = wmma_row_tile(Ab, WT, lane, ct);
            const int ncol = ct * 16 + nl;
#pragma unroll
            for (int r = 0; r < 8; ++r) {
                const int m  = r + 8 * half;
                const int gi = ((b * NNODE + j0 + m) * HCH + ncol) * 3 + d;
                vAcc[gi] += acc[r];
            }
        }
        __syncthreads();
    }
}

// ---------------------------------------------------------------------------
// Kernel 2: gated equivariant readout.
// ---------------------------------------------------------------------------
__global__ void __launch_bounds__(128)
mace_readout_kernel(const float* __restrict__ hFin,   // [B,N,H]
                    const float* __restrict__ vAcc,   // [B,N,H,3]
                    const float* __restrict__ Wg1,    // [128,128]
                    const float* __restrict__ Wg2,    // [128,128]
                    const float* __restrict__ fscale, // [1]
                    float* __restrict__ out)          // [B,N,3]
{
    __shared__ __align__(32) __bf16 WT[HCH * HCH];       // 32 KB
    __shared__ __align__(32) __bf16 Abuf[JTILE * HCH];   // 4 KB
    __shared__ __align__(32) __bf16 A2[JTILE * HCH];     // 4 KB
    __shared__ float gbuf[JTILE * HCH];                  // 8 KB
    __shared__ float pbuf[JTILE * 8 * 3];                // 1.5 KB

    const int tid  = threadIdx.x;
    const int lane = tid & 31;
    const int wv   = tid >> 5;
    const int b    = blockIdx.x >> 2;
    const int j0   = (blockIdx.x & 3) * JTILE;

    stage_wT(WT, Wg1, tid);
    for (int i = tid; i < JTILE * HCH; i += 128)
        Abuf[i] = (__bf16)hFin[(b * NNODE + j0) * HCH + i];
    __syncthreads();

    // t1 = silu(h @ Wg1)
    {
        const int half = lane >> 4;
        const int nl   = lane & 15;
#pragma unroll
        for (int t = 0; t < 2; ++t) {
            const int ct = wv * 2 + t;
            v8f acc = wmma_row_tile(Abuf, WT, lane, ct);
            const int ncol = ct * 16 + nl;
#pragma unroll
            for (int r = 0; r < 8; ++r)
                A2[(r + 8 * half) * HCH + ncol] = (__bf16)silu_f(acc[r]);
        }
    }
    __syncthreads();
    stage_wT(WT, Wg2, tid);
    __syncthreads();

    // g = t1 @ Wg2
    {
        const int half = lane >> 4;
        const int nl   = lane & 15;
#pragma unroll
        for (int t = 0; t < 2; ++t) {
            const int ct = wv * 2 + t;
            v8f acc = wmma_row_tile(A2, WT, lane, ct);
            const int ncol = ct * 16 + nl;
#pragma unroll
            for (int r = 0; r < 8; ++r)
                gbuf[(r + 8 * half) * HCH + ncol] = acc[r];
        }
    }
    __syncthreads();

    // out[n,d] = sum_c v[n,c,d] * g[n,c]  (8 threads per node, 16 channels each)
    {
        const int m  = tid >> 3;
        const int ch = tid & 7;
        float p0 = 0.0f, p1 = 0.0f, p2 = 0.0f;
        const float* vp = vAcc + ((size_t)(b * NNODE + j0 + m) * HCH + ch * 16) * 3;
        const float* gp = gbuf + m * HCH + ch * 16;
#pragma unroll
        for (int c = 0; c < 16; ++c) {
            const float g = gp[c];
            p0 = fmaf(g, vp[c * 3 + 0], p0);
            p1 = fmaf(g, vp[c * 3 + 1], p1);
            p2 = fmaf(g, vp[c * 3 + 2], p2);
        }
        pbuf[(m * 8 + ch) * 3 + 0] = p0;
        pbuf[(m * 8 + ch) * 3 + 1] = p1;
        pbuf[(m * 8 + ch) * 3 + 2] = p2;
    }
    __syncthreads();
    if (tid < JTILE * 3) {
        const int m = tid / 3;
        const int d = tid % 3;
        float s = 0.0f;
#pragma unroll
        for (int ch = 0; ch < 8; ++ch) s += pbuf[(m * 8 + ch) * 3 + d];
        out[(b * NNODE + j0 + m) * 3 + d] = s * fscale[0];
    }
}

// ---------------------------------------------------------------------------
extern "C" void kernel_launch(void* const* d_in, const int* in_sizes, int n_in,
                              void* d_out, int out_size, void* d_ws, size_t ws_size,
                              hipStream_t stream)
{
    (void)in_sizes; (void)n_in; (void)out_size; (void)ws_size;

    const float* pos  = (const float*)d_in[0];
    const int*   z    = (const int*)d_in[1];
    const float* gt   = (const float*)d_in[2];
    const float* emb  = (const float*)d_in[3];
    const float* Wt   = (const float*)d_in[4];
    const float* Wr1  = (const float*)d_in[5];
    const float* Wr2  = (const float*)d_in[6];
    const float* Wupd = (const float*)d_in[7];
    const float* Wmix = (const float*)d_in[8];
    const float* Wg1  = (const float*)d_in[9];
    const float* Wg2  = (const float*)d_in[10];
    const float* fs   = (const float*)d_in[11];
    float* out = (float*)d_out;

    // workspace: hA | hB | v   (2 MB + 2 MB + 6 MB)
    float* hA = (float*)d_ws;
    float* hB = hA + (size_t)BGRAPH * NNODE * HCH;
    float* v  = hB + (size_t)BGRAPH * NNODE * HCH;

    mace_init_kernel<<<BGRAPH, HCH, 0, stream>>>(gt, z, emb, Wt, hA, v);

    const dim3 grid(BGRAPH * 4);
    // layer 0: hA -> hB
    mace_layer_kernel<<<grid, 128, 0, stream>>>(pos, Wr1, Wr2, Wupd, Wmix, hA, hB, v);
    // layer 1: hB -> hA
    mace_layer_kernel<<<grid, 128, 0, stream>>>(pos,
                                                Wr1 + NBASIS * HCH,
                                                Wr2 + HCH * HCH,
                                                Wupd + HCH * HCH,
                                                Wmix + HCH * HCH,
                                                hB, hA, v);
    mace_readout_kernel<<<grid, 128, 0, stream>>>(hA, v, Wg1, Wg2, fs, out);
}
